// KernelAttention_17609365913759
// MI455X (gfx1250) — matrix operand order repaired and checked by reading, not compiled, not run on hardware
//
#include <hip/hip_runtime.h>
#include <hip/hip_bf16.h>

// ---------------- problem constants ----------------
#define B_     4
#define N_     2048
#define M_     512
#define H_     12
#define D_     64
#define DIM_   768
#define INNER_ 768
#define QKV3_  2304      // 3*INNER
#define SCALE_ 0.125f
#define NEGV_  (-1e9f)
#define TEMP_  24.0f

typedef __attribute__((ext_vector_type(16))) __bf16        v16bf;
typedef __attribute__((ext_vector_type(8)))  float         v8f;
typedef __attribute__((ext_vector_type(4)))  unsigned int  v4u;
typedef __attribute__((ext_vector_type(8)))  int           v8i;
typedef __attribute__((ext_vector_type(4)))  int           v4i;

union BFrag { v16bf v; v4u q[2]; unsigned short s[16]; __bf16 h[16]; };

// native bf16 convert (lowers to v_cvt_*bf16_f32 on gfx1250)
__device__ __forceinline__ unsigned short f2bf(float f) {
  __bf16 h = (__bf16)f;
  unsigned short u;
  __builtin_memcpy(&u, &h, 2);
  return u;
}
__device__ __forceinline__ float bf2f(unsigned short h) {
  union { unsigned u; float f; } v; v.u = ((unsigned)h) << 16; return v.f;
}

__device__ __forceinline__ v8f wmma_bf16(v16bf a, v16bf b, v8f c) {
  return __builtin_amdgcn_wmma_f32_16x16x32_bf16(false, a, false, b, (short)0, c,
                                                 false, false);
}

// A fragment (16x32 bf16, MxK): lane holds row (lane&15); per CDNA5 ISA layout
__device__ __forceinline__ v16bf load_a_frag_bf(const unsigned short* src, int ld,
                                                int rowBase, int kBase, int rows) {
  int lane = threadIdx.x & 31, half = lane >> 4, r = lane & 15;
  BFrag f;
  int row = rowBase + r;
  if (row < rows) {
    const unsigned short* p = src + (size_t)row * ld + kBase;
    f.q[0] = *(const v4u*)(p + half * 8);
    f.q[1] = *(const v4u*)(p + 16 + half * 8);
  } else {
    f.q[0] = (v4u)0u; f.q[1] = (v4u)0u;
  }
  return f.v;
}
__device__ __forceinline__ v16bf load_a_frag_f32(const float* src, int ld,
                                                 int rowBase, int kBase, int rows) {
  int lane = threadIdx.x & 31, half = lane >> 4, r = lane & 15;
  BFrag f;
  int row = rowBase + r;
  if (row < rows) {
    const float* p = src + (size_t)row * ld + kBase;
#pragma unroll
    for (int e = 0; e < 8; ++e) {
      f.h[e]     = (__bf16)p[half * 8 + e];
      f.h[8 + e] = (__bf16)p[16 + half * 8 + e];
    }
  } else {
    f.q[0] = (v4u)0u; f.q[1] = (v4u)0u;
  }
  return f.v;
}
// B fragment (32x16 bf16, KxN) from "N-major" storage src[n][k] (k contiguous)
__device__ __forceinline__ v16bf load_b_frag_bf(const unsigned short* src, int ld,
                                                int colBase, int kBase) {
  int lane = threadIdx.x & 31, half = lane >> 4, c = lane & 15;
  const unsigned short* p = src + (size_t)(colBase + c) * ld + kBase;
  BFrag f;
  f.q[0] = *(const v4u*)(p + half * 8);
  f.q[1] = *(const v4u*)(p + 16 + half * 8);
  return f.v;
}

// ---------------- Tensor Data Mover: 2-D tile global -> LDS ----------------
#if __has_builtin(__builtin_amdgcn_tensor_load_to_lds)
#define HAVE_TDM 1
__device__ __forceinline__ void tdm_load_2d(unsigned lds_off, const void* gptr,
                                            unsigned rowBytes, unsigned rows,
                                            unsigned strideBytes) {
  unsigned long long ga = (unsigned long long)(size_t)gptr;
  v4u g0;
  g0[0] = 1u;                                     // count=1, user descriptor
  g0[1] = lds_off;                                // lds_addr
  g0[2] = (unsigned)ga;                           // global_addr[31:0]
  g0[3] = (unsigned)((ga >> 32) & 0x1FFFFFFu) | (2u << 30);  // addr[56:32] | type=2
  v8i g1;
  g1[0] = 0;                                      // mask=0, data_size=1B, no flags
  g1[1] = (int)(rowBytes << 16);                  // tensor_dim0[15:0] (ab_addr=0)
  g1[2] = (int)((rows & 0xffffu) << 16) | (int)(rowBytes >> 16);  // td1 lo | td0 hi
  g1[3] = (int)(rowBytes << 16) | (int)(rows >> 16);              // tile_dim0 | td1 hi
  g1[4] = (int)rows;                              // tile_dim1 (tile_dim2 = 0)
  g1[5] = (int)strideBytes;                       // tensor_dim0_stride[31:0]
  g1[6] = 0;
  g1[7] = 0;
  v4i z4 = {0, 0, 0, 0};
#if __clang_major__ >= 23
  v8i z8 = {0, 0, 0, 0, 0, 0, 0, 0};
  __builtin_amdgcn_tensor_load_to_lds(g0, g1, z4, z4, z8, 0);
#else
  __builtin_amdgcn_tensor_load_to_lds(g0, g1, z4, z4, 0);
#endif
}
#endif

// ---------------- kernel 1: W -> bf16 transpose  WT[c][k] = W[k][c] ----------------
__global__ void kT_w_transpose(const float* __restrict__ W, int K, int C,
                               unsigned short* __restrict__ WT) {
  int idx = blockIdx.x * blockDim.x + threadIdx.x;
  if (idx >= K * C) return;
  int k = idx / C, c = idx % C;
  WT[(size_t)c * K + k] = f2bf(W[idx]);
}

// ---------------- kernel 2: QKV GEMM, one wave per 16x64 tile ----------------
__global__ __launch_bounds__(32) void kT_qkv_gemm(
    const float* __restrict__ X, int rows, int seqLen,
    const unsigned short* __restrict__ WT,   // [2304][768] bf16
    unsigned short* __restrict__ oQ, unsigned short* __restrict__ oK,
    unsigned short* __restrict__ oVT) {
  int rowBase = blockIdx.x * 16, colBase = blockIdx.y * 64;
  v8f a0 = {0.f,0.f,0.f,0.f,0.f,0.f,0.f,0.f}, a1 = a0, a2 = a0, a3 = a0;
#pragma unroll 2
  for (int k = 0; k < DIM_; k += 32) {
    v16bf a = load_a_frag_f32(X, DIM_, rowBase, k, rows);
    a0 = wmma_bf16(a, load_b_frag_bf(WT, DIM_, colBase,      k), a0);
    a1 = wmma_bf16(a, load_b_frag_bf(WT, DIM_, colBase + 16, k), a1);
    a2 = wmma_bf16(a, load_b_frag_bf(WT, DIM_, colBase + 32, k), a2);
    a3 = wmma_bf16(a, load_b_frag_bf(WT, DIM_, colBase + 48, k), a3);
  }
  int lane = threadIdx.x & 31, half = lane >> 4, r = lane & 15;
  int which = colBase / INNER_;
  int h = (colBase % INNER_) / D_;          // 64-col group == one head slab
  v8f accs[4] = {a0, a1, a2, a3};
#pragma unroll
  for (int cc = 0; cc < 4; ++cc) {
    int d = cc * 16 + r;
#pragma unroll
    for (int t = 0; t < 8; ++t) {
      int row = rowBase + half * 8 + t;
      if (row >= rows) continue;
      int bi = row / seqLen, n = row % seqLen;
      unsigned short v = f2bf(accs[cc][t]);
      if (which == 0)
        oQ[(((size_t)bi * H_ + h) * seqLen + n) * D_ + d] = v;
      else if (which == 1)
        oK[(((size_t)bi * H_ + h) * seqLen + n) * D_ + d] = v;
      else
        oVT[(((size_t)bi * H_ + h) * D_ + d) * seqLen + n] = v;
    }
  }
}

// ---------------- kernel 3: branch 1 (instance queries vs kernel keys) ----------------
__global__ __launch_bounds__(128) void kT_branch1(
    const unsigned short* __restrict__ tq, const unsigned short* __restrict__ kk,
    const unsigned short* __restrict__ kvT, const float* __restrict__ krd,
    const unsigned char* __restrict__ mask, float* __restrict__ out4,
    unsigned short* __restrict__ att_ws) {
  extern __shared__ __align__(16) char smem[];
  const int SD = M_ + 8;                       // 520, pad vs bank conflicts
  float* sdots     = (float*)smem;             // [16][SD]
  float* rowmax    = sdots + 16 * SD;
  float* rowsum    = rowmax + 16;
  float* rowsumhot = rowsum + 16;
  float* pacc      = rowsumhot + 16;           // [4][16][64]; doubles as red scratch
  unsigned short* lvb = (unsigned short*)(pacc + 4 * 16 * 64);   // [64][512] bf16

  int n0 = blockIdx.x * 16, h = blockIdx.y, b = blockIdx.z;
  int bh = b * H_ + h;
  int lane = threadIdx.x & 31, wave = threadIdx.x >> 5;
  int half = lane >> 4, r = lane & 15;

  const unsigned short* qb = tq + ((size_t)bh * N_ + n0) * D_;
  const unsigned short* kb = kk + (size_t)bh * M_ * D_;
  const unsigned short* vb = kvT + (size_t)bh * D_ * M_;
  const unsigned char*  mb = mask + (size_t)bh * N_ * M_;
  const float*          kr = krd + (size_t)bh * M_ * N_;

  // async: DMA the whole 64x512 bf16 V^T slab into LDS while we do Q.K^T
#ifdef HAVE_TDM
  if (wave == 0)
    tdm_load_2d((unsigned)(size_t)(void*)lvb, vb, M_ * 2, D_, M_ * 2);
#else
  for (int i = threadIdx.x; i < D_ * M_ / 8; i += 128)
    ((v4u*)lvb)[i] = ((const v4u*)vb)[i];
#endif

  v16bf a0 = load_a_frag_bf(qb, D_, 0, 0, 1 << 30);
  v16bf a1 = load_a_frag_bf(qb, D_, 0, 32, 1 << 30);

  // phase A: dots -> LDS (each wave: 8 col tiles)
  for (int j = wave * 8; j < wave * 8 + 8; ++j) {
    v8f c = {0.f,0.f,0.f,0.f,0.f,0.f,0.f,0.f};
    c = wmma_bf16(a0, load_b_frag_bf(kb, D_, j * 16, 0), c);
    c = wmma_bf16(a1, load_b_frag_bf(kb, D_, j * 16, 32), c);
    int m = j * 16 + r;
#pragma unroll
    for (int t = 0; t < 8; ++t) {
      int rr = half * 8 + t;
      float s = c[t] * SCALE_;
      if (mb[(size_t)(n0 + rr) * M_ + m]) s = NEGV_;
      sdots[rr * SD + m] = s;
    }
  }
  __syncthreads();

  // phase B: parallel row max + dual softmax denominators (8 strips/row)
  {
    int rr = threadIdx.x >> 3, st = threadIdx.x & 7;
    const int CW = M_ / 8;
    float mx = -3.4e38f;
    for (int m = st * CW; m < (st + 1) * CW; ++m)
      mx = fmaxf(mx, sdots[rr * SD + m]);
    pacc[rr * 8 + st] = mx;
    __syncthreads();
    if (threadIdx.x < 16) {
      float m2 = pacc[threadIdx.x * 8];
#pragma unroll
      for (int u = 1; u < 8; ++u) m2 = fmaxf(m2, pacc[threadIdx.x * 8 + u]);
      rowmax[threadIdx.x] = m2;
    }
    __syncthreads();
    float rmx = rowmax[rr], s1 = 0.f, s2 = 0.f;
    for (int m = st * CW; m < (st + 1) * CW; ++m) {
      float d = sdots[rr * SD + m] - rmx;
      s1 += __expf(d);
      s2 += __expf(TEMP_ * d);
    }
    pacc[128 + rr * 8 + st] = s1;
    pacc[256 + rr * 8 + st] = s2;
    __syncthreads();
    if (threadIdx.x < 16) {
      float t1 = 0.f, t2 = 0.f;
#pragma unroll
      for (int u = 0; u < 8; ++u) {
        t1 += pacc[128 + threadIdx.x * 8 + u];
        t2 += pacc[256 + threadIdx.x * 8 + u];
      }
      rowsum[threadIdx.x] = t1; rowsumhot[threadIdx.x] = t2;
    }
  }
#ifdef HAVE_TDM
  if (wave == 0) __builtin_amdgcn_s_wait_tensorcnt(0);   // V slab resident
#endif
  __syncthreads();

  // phase C: attn_hot output (already transposed: out4[b,h,m,n])
  float* o4 = out4 + (size_t)bh * M_ * N_;
  for (int m = threadIdx.x; m < M_; m += 128) {
    if (m + 128 < M_) __builtin_prefetch(kr + (size_t)(m + 128) * N_ + n0);
#pragma unroll
    for (int nn = 0; nn < 16; ++nn) {
      float d = sdots[nn * SD + m] - rowmax[nn];
      float hot = __expf(TEMP_ * d) / rowsumhot[nn];
      o4[(size_t)m * N_ + n0 + nn] = hot * kr[(size_t)m * N_ + n0 + nn];
    }
  }

  // phase D: P @ V from LDS-resident V (each wave: 128 keys)
  v8f acc0 = {0.f,0.f,0.f,0.f,0.f,0.f,0.f,0.f}, acc1 = acc0, acc2 = acc0, acc3 = acc0;
  float invs = 1.0f / rowsum[r], mx = rowmax[r];
  for (int c0 = wave * 128; c0 < wave * 128 + 128; c0 += 32) {
    BFrag pf;
#pragma unroll
    for (int e = 0; e < 8; ++e) {
      int m = c0 + half * 8 + e;
      pf.h[e] = (__bf16)(__expf(sdots[r * SD + m] - mx) * invs *
                         kr[(size_t)m * N_ + n0 + r]);
      int m2 = c0 + 16 + half * 8 + e;
      pf.h[8 + e] = (__bf16)(__expf(sdots[r * SD + m2] - mx) * invs *
                             kr[(size_t)m2 * N_ + n0 + r]);
    }
    acc0 = wmma_bf16(pf.v, load_b_frag_bf(lvb, M_, 0,  c0), acc0);
    acc1 = wmma_bf16(pf.v, load_b_frag_bf(lvb, M_, 16, c0), acc1);
    acc2 = wmma_bf16(pf.v, load_b_frag_bf(lvb, M_, 32, c0), acc2);
    acc3 = wmma_bf16(pf.v, load_b_frag_bf(lvb, M_, 48, c0), acc3);
  }
#pragma unroll
  for (int t = 0; t < 8; ++t) {
    int rr = half * 8 + t;
    pacc[(wave * 16 + rr) * 64 + r]      = acc0[t];
    pacc[(wave * 16 + rr) * 64 + 16 + r] = acc1[t];
    pacc[(wave * 16 + rr) * 64 + 32 + r] = acc2[t];
    pacc[(wave * 16 + rr) * 64 + 48 + r] = acc3[t];
  }
  __syncthreads();
#pragma unroll
  for (int t = 0; t < 8; ++t) {
    int idx = threadIdx.x * 8 + t;
    int rr = idx >> 6, d = idx & 63;
    float s = pacc[rr * 64 + d] + pacc[(16 + rr) * 64 + d] +
              pacc[(32 + rr) * 64 + d] + pacc[(48 + rr) * 64 + d];
    att_ws[((size_t)b * N_ + n0 + rr) * INNER_ + h * D_ + d] = f2bf(s);
  }
}

// ---------------- kernel 4: branch 2 (kernel queries vs instance keys) ----------------
__global__ __launch_bounds__(128) void kT_branch2(
    const unsigned short* __restrict__ kq, const unsigned short* __restrict__ tk,
    const unsigned short* __restrict__ tvT, const float* __restrict__ krd,
    const unsigned char* __restrict__ mask, unsigned short* __restrict__ k_ws) {
  extern __shared__ __align__(16) char smem[];
  const int SD = N_ + 8;                       // 2056
  float* sdots  = (float*)smem;                // [16][SD]
  float* rowmax = sdots + 16 * SD;
  float* rowsum = rowmax + 16;
  float* pacc   = rowsum + 16;                 // [4][16][64]; doubles as red scratch
  unsigned short* vbuf = (unsigned short*)(pacc + 4 * 16 * 64);  // [4][2][64*32] bf16

  int m0 = blockIdx.x * 16, h = blockIdx.y, b = blockIdx.z;
  int bh = b * H_ + h;
  int lane = threadIdx.x & 31, wave = threadIdx.x >> 5;
  int half = lane >> 4, r = lane & 15;

  const unsigned short* qb = kq + ((size_t)bh * M_ + m0) * D_;
  const unsigned short* kb = tk + (size_t)bh * N_ * D_;
  const unsigned short* vb = tvT + (size_t)bh * D_ * N_;
  const unsigned char*  mb = mask + (size_t)bh * N_ * M_;
  const float*          kr = krd + (size_t)bh * M_ * N_;

  unsigned short* myb0 = vbuf + wave * 2 * (D_ * 32);
  unsigned short* myb1 = myb0 + D_ * 32;

#ifdef HAVE_TDM
  // prefetch V chunk 0 (strided 64x32 tile) while we compute the dots
  tdm_load_2d((unsigned)(size_t)(void*)myb0, vb + wave * 512, 64u, (unsigned)D_,
              (unsigned)(N_ * 2));
#endif

  v16bf a0 = load_a_frag_bf(qb, D_, 0, 0, 1 << 30);
  v16bf a1 = load_a_frag_bf(qb, D_, 0, 32, 1 << 30);

  for (int j = wave * 32; j < wave * 32 + 32; ++j) {
    v8f c = {0.f,0.f,0.f,0.f,0.f,0.f,0.f,0.f};
    c = wmma_bf16(a0, load_b_frag_bf(kb, D_, j * 16, 0), c);
    c = wmma_bf16(a1, load_b_frag_bf(kb, D_, j * 16, 32), c);
    int n = j * 16 + r;
#pragma unroll
    for (int t = 0; t < 8; ++t) {
      int rr = half * 8 + t;
      float s = c[t] * SCALE_;
      if (mb[(size_t)n * M_ + m0 + rr]) s = NEGV_;   // mask transposed
      sdots[rr * SD + n] = s;
    }
  }
  __syncthreads();

  // parallel row max + softmax denominator (8 strips per row)
  {
    int rr = threadIdx.x >> 3, st = threadIdx.x & 7;
    const int CW = N_ / 8;
    float mx = -3.4e38f;
    for (int n = st * CW; n < (st + 1) * CW; ++n)
      mx = fmaxf(mx, sdots[rr * SD + n]);
    pacc[rr * 8 + st] = mx;
    __syncthreads();
    if (threadIdx.x < 16) {
      float m2 = pacc[threadIdx.x * 8];
#pragma unroll
      for (int u = 1; u < 8; ++u) m2 = fmaxf(m2, pacc[threadIdx.x * 8 + u]);
      rowmax[threadIdx.x] = m2;
    }
    __syncthreads();
    float rmx = rowmax[rr], s1 = 0.f;
    for (int n = st * CW; n < (st + 1) * CW; ++n)
      s1 += __expf(sdots[rr * SD + n] - rmx);
    pacc[128 + rr * 8 + st] = s1;
    __syncthreads();
    if (threadIdx.x < 16) {
      float t1 = 0.f;
#pragma unroll
      for (int u = 0; u < 8; ++u) t1 += pacc[128 + threadIdx.x * 8 + u];
      rowsum[threadIdx.x] = t1;
    }
  }
  __syncthreads();

  v8f acc0 = {0.f,0.f,0.f,0.f,0.f,0.f,0.f,0.f}, acc1 = acc0, acc2 = acc0, acc3 = acc0;
  float invs = 1.0f / rowsum[r], mx = rowmax[r];
  const float* krow = kr + (size_t)(m0 + r) * N_;
  // P @ V: 16 chunks of 32 keys, TDM double-buffered per wave
  for (int i = 0; i < 16; ++i) {
    int c0 = wave * 512 + i * 32;
    unsigned short* cur = (i & 1) ? myb1 : myb0;
#ifdef HAVE_TDM
    if (i + 1 < 16) {
      unsigned short* nxt = (i & 1) ? myb0 : myb1;
      tdm_load_2d((unsigned)(size_t)(void*)nxt, vb + c0 + 32, 64u, (unsigned)D_,
                  (unsigned)(N_ * 2));
      __builtin_amdgcn_s_wait_tensorcnt(1);   // chunk i resident (in-order)
    } else {
      __builtin_amdgcn_s_wait_tensorcnt(0);
    }
#else
    for (int u = threadIdx.x & 31; u < D_ * 32 / 8; u += 32)
      ((v4u*)cur)[u] = ((const v4u*)(vb))[(u * 8 / 32) * (N_ / 8) + (c0 / 8) + ((u * 8) % 32) / 8];
#endif
    __builtin_prefetch(krow + c0 + 32);
    BFrag pf;
#pragma unroll
    for (int e = 0; e < 8; ++e) {
      int n = c0 + half * 8 + e;
      pf.h[e] = (__bf16)(__expf(sdots[r * SD + n] - mx) * invs * krow[n]);
      int n2 = c0 + 16 + half * 8 + e;
      pf.h[8 + e] = (__bf16)(__expf(sdots[r * SD + n2] - mx) * invs * krow[n2]);
    }
#ifdef HAVE_TDM
    acc0 = wmma_bf16(pf.v, load_b_frag_bf(cur, 32, 0,  0), acc0);
    acc1 = wmma_bf16(pf.v, load_b_frag_bf(cur, 32, 16, 0), acc1);
    acc2 = wmma_bf16(pf.v, load_b_frag_bf(cur, 32, 32, 0), acc2);
    acc3 = wmma_bf16(pf.v, load_b_frag_bf(cur, 32, 48, 0), acc3);
#else
    acc0 = wmma_bf16(pf.v, load_b_frag_bf(vb, N_, 0,  c0), acc0);
    acc1 = wmma_bf16(pf.v, load_b_frag_bf(vb, N_, 16, c0), acc1);
    acc2 = wmma_bf16(pf.v, load_b_frag_bf(vb, N_, 32, c0), acc2);
    acc3 = wmma_bf16(pf.v, load_b_frag_bf(vb, N_, 48, c0), acc3);
#endif
  }
#pragma unroll
  for (int t = 0; t < 8; ++t) {
    int rr = half * 8 + t;
    pacc[(wave * 16 + rr) * 64 + r]      = acc0[t];
    pacc[(wave * 16 + rr) * 64 + 16 + r] = acc1[t];
    pacc[(wave * 16 + rr) * 64 + 32 + r] = acc2[t];
    pacc[(wave * 16 + rr) * 64 + 48 + r] = acc3[t];
  }
  __syncthreads();
#pragma unroll
  for (int t = 0; t < 8; ++t) {
    int idx = threadIdx.x * 8 + t;
    int rr = idx >> 6, d = idx & 63;
    float s = pacc[rr * 64 + d] + pacc[(16 + rr) * 64 + d] +
              pacc[(32 + rr) * 64 + d] + pacc[(48 + rr) * 64 + d];
    k_ws[((size_t)b * M_ + m0 + rr) * INNER_ + h * D_ + d] = f2bf(s);
  }
}

// ---------------- kernel 5: branch 3 (cluster query), tiny ----------------
__global__ __launch_bounds__(64) void kT_branch3(
    const unsigned short* __restrict__ cq, const unsigned short* __restrict__ kk,
    const unsigned short* __restrict__ kvT, const unsigned char* __restrict__ mask,
    unsigned short* __restrict__ c_ws) {
  __shared__ float q[D_];
  __shared__ float s[M_];
  __shared__ float red[2];
  int h = blockIdx.x, b = blockIdx.y, bh = b * H_ + h;
  int t = threadIdx.x;
  q[t] = bf2f(cq[(size_t)bh * D_ + t]);
  __syncthreads();
  for (int m = t; m < M_; m += 64) {
    float dot = 0.f;
    const unsigned short* kr = kk + ((size_t)bh * M_ + m) * D_;
#pragma unroll 8
    for (int d = 0; d < D_; ++d) dot += q[d] * bf2f(kr[d]);
    float v = dot * SCALE_;
    if (mask[(size_t)bh * N_ * M_ + m]) v = NEGV_;   // att_mask[:, :, :1]
    s[m] = v;
  }
  __syncthreads();
  if (t == 0) {
    float mx = -3.4e38f;
    for (int m = 0; m < M_; ++m) mx = fmaxf(mx, s[m]);
    float sm = 0.f;
    for (int m = 0; m < M_; ++m) sm += __expf(s[m] - mx);
    red[0] = mx; red[1] = sm;
  }
  __syncthreads();
  float mx = red[0], inv = 1.0f / red[1];
  for (int m = t; m < M_; m += 64) s[m] = __expf(s[m] - mx) * inv;
  __syncthreads();
  {
    int d = t;
    float acc = 0.f;
    const unsigned short* vr = kvT + ((size_t)bh * D_ + d) * M_;
    for (int m = 0; m < M_; ++m) acc += s[m] * bf2f(vr[m]);
    c_ws[(size_t)b * INNER_ + h * D_ + d] = f2bf(acc);
  }
}

// ---------------- kernel 6: output projection GEMM (16x64 per wave) ----------------
__global__ __launch_bounds__(32) void kT_proj_gemm(
    const unsigned short* __restrict__ A, int rows,
    const unsigned short* __restrict__ WT,   // [768][768] bf16 (W_out^T)
    const float* __restrict__ bias, float* __restrict__ Out) {
  int rowBase = blockIdx.x * 16, colBase = blockIdx.y * 64;
  v8f a0 = {0.f,0.f,0.f,0.f,0.f,0.f,0.f,0.f}, a1 = a0, a2 = a0, a3 = a0;
#pragma unroll 2
  for (int k = 0; k < INNER_; k += 32) {
    v16bf a = load_a_frag_bf(A, INNER_, rowBase, k, rows);
    a0 = wmma_bf16(a, load_b_frag_bf(WT, INNER_, colBase,      k), a0);
    a1 = wmma_bf16(a, load_b_frag_bf(WT, INNER_, colBase + 16, k), a1);
    a2 = wmma_bf16(a, load_b_frag_bf(WT, INNER_, colBase + 32, k), a2);
    a3 = wmma_bf16(a, load_b_frag_bf(WT, INNER_, colBase + 48, k), a3);
  }
  int lane = threadIdx.x & 31, half = lane >> 4, r = lane & 15;
  v8f accs[4] = {a0, a1, a2, a3};
#pragma unroll
  for (int cc = 0; cc < 4; ++cc) {
    int col = colBase + cc * 16 + r;
    float bz = bias[col];
#pragma unroll
    for (int t = 0; t < 8; ++t) {
      int row = rowBase + half * 8 + t;
      if (row < rows) Out[(size_t)row * DIM_ + col] = accs[cc][t] + bz;
    }
  }
}

// ---------------- host-side launch ----------------
extern "C" void kernel_launch(void* const* d_in, const int* in_sizes, int n_in,
                              void* d_out, int out_size, void* d_ws, size_t ws_size,
                              hipStream_t stream) {
  const float*         x     = (const float*)d_in[0];
  const float*         kx    = (const float*)d_in[1];
  const float*         krd   = (const float*)d_in[2];
  const float*         clst  = (const float*)d_in[3];
  const unsigned char* amask = (const unsigned char*)d_in[4];  // jax bool -> 1 byte
  const float*         Wqkv  = (const float*)d_in[5];
  const float*         Wout  = (const float*)d_in[6];
  const float*         bout  = (const float*)d_in[7];
  (void)in_sizes; (void)n_in; (void)out_size; (void)ws_size;

  float* out = (float*)d_out;
  float* out1 = out;                                        // [B,N,DIM]
  float* out2 = out1 + (size_t)B_ * N_ * DIM_;              // [B,M,DIM]
  float* out3 = out2 + (size_t)B_ * M_ * DIM_;              // [B,1,DIM]
  float* out4 = out3 + (size_t)B_ * DIM_;                   // [B,H,M,N]

  char* ws = (char*)d_ws;
  size_t off = 0;
  auto take = [&](size_t bytes) { char* p = ws + off; off = (off + bytes + 255) & ~(size_t)255; return p; };
  unsigned short* wqkvT = (unsigned short*)take((size_t)QKV3_ * DIM_ * 2);
  unsigned short* woutT = (unsigned short*)take((size_t)DIM_ * INNER_ * 2);
  unsigned short* tq  = (unsigned short*)take((size_t)B_ * H_ * N_ * D_ * 2);
  unsigned short* tk  = (unsigned short*)take((size_t)B_ * H_ * N_ * D_ * 2);
  unsigned short* tvT = (unsigned short*)take((size_t)B_ * H_ * N_ * D_ * 2);
  unsigned short* kq  = (unsigned short*)take((size_t)B_ * H_ * M_ * D_ * 2);
  unsigned short* kk  = (unsigned short*)take((size_t)B_ * H_ * M_ * D_ * 2);
  unsigned short* kvT = (unsigned short*)take((size_t)B_ * H_ * M_ * D_ * 2);
  unsigned short* cq  = (unsigned short*)take((size_t)B_ * H_ * D_ * 2);
  unsigned short* ck  = (unsigned short*)take((size_t)B_ * H_ * D_ * 2);
  unsigned short* cvT = (unsigned short*)take((size_t)B_ * H_ * D_ * 2);
  unsigned short* att_ws = (unsigned short*)take((size_t)B_ * N_ * INNER_ * 2);
  unsigned short* k_ws   = (unsigned short*)take((size_t)B_ * M_ * INNER_ * 2);
  unsigned short* c_ws   = (unsigned short*)take((size_t)B_ * INNER_ * 2);

  {
    int tot = DIM_ * QKV3_;
    kT_w_transpose<<<(tot + 255) / 256, 256, 0, stream>>>(Wqkv, DIM_, QKV3_, wqkvT);
    tot = INNER_ * DIM_;
    kT_w_transpose<<<(tot + 255) / 256, 256, 0, stream>>>(Wout, INNER_, DIM_, woutT);
  }
  kT_qkv_gemm<<<dim3((B_ * N_) / 16, QKV3_ / 64), 32, 0, stream>>>(x, B_ * N_, N_, wqkvT, tq, tk, tvT);
  kT_qkv_gemm<<<dim3((B_ * M_) / 16, QKV3_ / 64), 32, 0, stream>>>(kx, B_ * M_, M_, wqkvT, kq, kk, kvT);
  kT_qkv_gemm<<<dim3(1, QKV3_ / 64), 32, 0, stream>>>(clst, B_, 1, wqkvT, cq, ck, cvT);

  size_t smem1 = (16 * (M_ + 8) + 48 + 4 * 16 * 64) * sizeof(float)
               + (size_t)D_ * M_ * 2;                                  // + V slab (~113 KB)
  kT_branch1<<<dim3(N_ / 16, H_, B_), 128, smem1, stream>>>(tq, kk, kvT, krd, amask, out4, att_ws);
  size_t smem2 = (16 * (N_ + 8) + 32 + 4 * 16 * 64) * sizeof(float)
               + (size_t)4 * 2 * D_ * 32 * 2;                          // + V chunk bufs (~181 KB)
  kT_branch2<<<dim3(M_ / 16, H_, B_), 128, smem2, stream>>>(kq, tk, tvT, krd, amask, k_ws);
  kT_branch3<<<dim3(H_, B_), 64, 0, stream>>>(cq, kk, kvT, amask, c_ws);

  kT_proj_gemm<<<dim3((B_ * N_) / 16, DIM_ / 64), 32, 0, stream>>>(att_ws, B_ * N_, woutT, bout, out1);
  kT_proj_gemm<<<dim3((B_ * M_) / 16, DIM_ / 64), 32, 0, stream>>>(k_ws, B_ * M_, woutT, bout, out2);
  kT_proj_gemm<<<dim3(1, DIM_ / 64), 32, 0, stream>>>(c_ws, B_, woutT, bout, out3);
}
